// Attention_15161234555039
// MI455X (gfx1250) — compile-verified, matrix operand order checked by reading, not run on hardware
//
#include <hip/hip_runtime.h>
#include <hip/hip_bf16.h>

// ---------------- problem constants ----------------
#define BATCH   32
#define NTOK    513
#define NPAD    544            // 17 * 32, padded key/query dim
#define CDIM    768
#define THREEC  2304
#define NH      12
#define HDIM    64
#define MROWS   (BATCH * NTOK) // 16416 (multiple of 32)
#define QTILES  17             // ceil(513/32)
#define SCALE   0.125f         // 64^-0.5
#define EPSC    1e-6f
#define EPSN    (1e-6f / 513.0f)
#define NKEY    512            // N - NUM_EXTRA
#define NLEFT   359

// ---------------- WMMA types ----------------
typedef __attribute__((ext_vector_type(16))) __bf16 bf16x16;
typedef __attribute__((ext_vector_type(8)))  __bf16 bf16x8;
typedef __attribute__((ext_vector_type(8)))  float  f32x8;

__device__ __forceinline__ f32x8 zero8() {
  f32x8 v;
#pragma unroll
  for (int i = 0; i < 8; ++i) v[i] = 0.0f;
  return v;
}

// Load one 16x32 (A) or 32x16 (B) bf16 fragment.  Per the CDNA5 layout,
// lane L<16 holds K={0..7,16..23}, lane L>=16 holds K={8..15,24..31},
// so each lane does two contiguous 16-byte loads from its row.
__device__ __forceinline__ bf16x16 load_frag(const unsigned short* p) {
  bf16x8 lo = *reinterpret_cast<const bf16x8*>(p);
  bf16x8 hi = *reinterpret_cast<const bf16x8*>(p + 16);
  return __builtin_shufflevector(lo, hi, 0,1,2,3,4,5,6,7,8,9,10,11,12,13,14,15);
}

__device__ __forceinline__ f32x8 wmma_bf16(bf16x16 a, bf16x16 b, f32x8 c) {
  return __builtin_amdgcn_wmma_f32_16x16x32_bf16(false, a, false, b, (short)0, c,
                                                 false, false);
}

__device__ __forceinline__ unsigned short f32_bf16(float f) {
  unsigned int u = __float_as_uint(f);
  u += 0x7fffu + ((u >> 16) & 1u);       // round-to-nearest-even
  return (unsigned short)(u >> 16);
}
__device__ __forceinline__ float bf16_f32(unsigned short h) {
  return __uint_as_float(((unsigned int)h) << 16);
}

// ---------------- utility kernels ----------------
__global__ void kzero(float4* p, long long n) {
  long long i = (long long)blockIdx.x * blockDim.x + threadIdx.x;
  long long s = (long long)gridDim.x * blockDim.x;
  float4 z = make_float4(0.f, 0.f, 0.f, 0.f);
  for (; i < n; i += s) p[i] = z;
}

__global__ void kconv(const float* __restrict__ in, unsigned short* __restrict__ out,
                      int n) {
  int i = blockIdx.x * blockDim.x + threadIdx.x;
  int s = gridDim.x * blockDim.x;
  for (; i < n; i += s) out[i] = f32_bf16(in[i]);
}

// ---------------- qkv GEMM: [16416,768] x [768,2304]^T ----------------
// One wave computes a 32x64 output tile (2x4 WMMA fragments, 8 f32x8 accs),
// scatters q (scaled) / k into [b,h,tok,d] and v transposed into [b,h,d,tok].
__global__ __launch_bounds__(32)
void kqkv(const unsigned short* __restrict__ xbf,
          const unsigned short* __restrict__ wbf,
          unsigned short* __restrict__ qb,
          unsigned short* __restrict__ kb,
          unsigned short* __restrict__ vtb) {
  const int lane = threadIdx.x;
  const int r15  = lane & 15;
  const int off  = (lane >> 4) * 8;
  const int m0 = blockIdx.x * 32;
  const int n0 = blockIdx.y * 64;

  const unsigned short* a0 = xbf + (m0 + r15) * CDIM;
  const unsigned short* a1 = a0 + 16 * CDIM;
  const unsigned short* bp = wbf + (n0 + r15) * CDIM;

  f32x8 acc[8];
#pragma unroll
  for (int i = 0; i < 8; ++i) acc[i] = zero8();

#pragma unroll 2
  for (int k0 = 0; k0 < CDIM; k0 += 32) {
    bf16x16 A0 = load_frag(a0 + k0 + off);
    bf16x16 A1 = load_frag(a1 + k0 + off);
#pragma unroll
    for (int nt = 0; nt < 4; ++nt) {
      bf16x16 B = load_frag(bp + nt * 16 * CDIM + k0 + off);
      acc[nt]     = wmma_bf16(A0, B, acc[nt]);
      acc[4 + nt] = wmma_bf16(A1, B, acc[4 + nt]);
    }
  }

  const int rhi = 8 * (lane >> 4);
#pragma unroll
  for (int mt = 0; mt < 2; ++mt) {
#pragma unroll
    for (int nt = 0; nt < 4; ++nt) {
      const f32x8& a = acc[mt * 4 + nt];
#pragma unroll
      for (int r = 0; r < 8; ++r) {
        int m = m0 + mt * 16 + r + rhi;
        int n = n0 + nt * 16 + r15;
        int b   = m / NTOK;
        int tok = m - b * NTOK;
        int which = n / CDIM;
        int c = n - which * CDIM;
        int h = c >> 6, d = c & 63;
        float v = a[r];
        if (which == 0)
          qb[((b * NH + h) * NPAD + tok) * HDIM + d] = f32_bf16(v * SCALE);
        else if (which == 1)
          kb[((b * NH + h) * NPAD + tok) * HDIM + d] = f32_bf16(v);
        else
          vtb[((b * NH + h) * HDIM + d) * NPAD + tok] = f32_bf16(v);
      }
    }
  }
}

// ---------------- attention: per (b,h,32-query tile) ----------------
__global__ __launch_bounds__(128)
void kattn(const unsigned short* __restrict__ qb,
           const unsigned short* __restrict__ kb,
           const unsigned short* __restrict__ vtb,
           const float* __restrict__ mask,
           unsigned short* __restrict__ ob,
           float* __restrict__ score) {
  extern __shared__ char smem[];
  float*          S  = (float*)smem;                            // [32][NPAD] f32
  unsigned short* P  = (unsigned short*)(smem + 32 * NPAD * 4); // [32][NPAD] bf16
  unsigned short* Qs = (unsigned short*)(smem + 32 * NPAD * 4 + 32 * NPAD * 2); // [32][64]

  int bid = blockIdx.x;
  int b   = bid / (NH * QTILES);
  int rem = bid - b * (NH * QTILES);
  int h   = rem / QTILES;
  int qt  = rem - h * QTILES;
  int q0  = qt * 32;

  const int tid  = threadIdx.x;
  const int wave = tid >> 5;
  const int lane = tid & 31;
  const int r15  = lane & 15;
  const int off  = (lane >> 4) * 8;
  const int rhi  = 8 * (lane >> 4);

  const unsigned short* qbase = qb  + (b * NH + h) * NPAD * HDIM;
  const unsigned short* kbase = kb  + (b * NH + h) * NPAD * HDIM;
  const unsigned short* vbase = vtb + (b * NH + h) * HDIM * NPAD;

  // ---- async-stage the 32x64 Q tile into LDS (shared by all 4 waves) ----
  // 32 rows * 128B = 256 16-byte chunks; 128 threads issue 2 async b128 each.
  // LDS byte offset = low 32 bits of the generic pointer (flat->LDS aperture).
#pragma unroll
  for (int c2 = 0; c2 < 2; ++c2) {
    int chunk = tid + c2 * 128;          // 0..255
    int row   = chunk >> 3;              // 8 chunks per 128B row
    int c16   = chunk & 7;
    const unsigned short* gsrc = qbase + (q0 + row) * HDIM + c16 * 8;
    unsigned int ldsOff = (unsigned int)(unsigned long long)(Qs + chunk * 8);
    asm volatile("global_load_async_to_lds_b128 %0, %1, off"
                 :: "v"(ldsOff), "v"(gsrc) : "memory");
  }
  asm volatile("s_wait_asynccnt 0x0" ::: "memory");
  __syncthreads();

  // Q fragments for both 16-row halves and both K-steps of d (64 = 2*32)
  bf16x16 QA[2][2];
#pragma unroll
  for (int rh = 0; rh < 2; ++rh)
#pragma unroll
    for (int ks = 0; ks < 2; ++ks)
      QA[rh][ks] = load_frag(Qs + (rh * 16 + r15) * HDIM + ks * 32 + off);

  // ---- S = Q @ K^T (waves split the 34 key tiles) ----
  for (int kt = wave; kt < NPAD / 16; kt += 4) {
    int key0 = kt * 16;
    bf16x16 KB0 = load_frag(kbase + (key0 + r15) * HDIM + off);
    bf16x16 KB1 = load_frag(kbase + (key0 + r15) * HDIM + 32 + off);
    f32x8 s0 = zero8(), s1 = zero8();
    s0 = wmma_bf16(QA[0][0], KB0, s0);
    s0 = wmma_bf16(QA[0][1], KB1, s0);
    s1 = wmma_bf16(QA[1][0], KB0, s1);
    s1 = wmma_bf16(QA[1][1], KB1, s1);
    int col = key0 + r15;
    bool valid = col < NTOK;
#pragma unroll
    for (int r = 0; r < 8; ++r) {
      int row = r + rhi;
      S[row * NPAD + col]        = valid ? s0[r] : -1e30f;
      S[(row + 16) * NPAD + col] = valid ? s1[r] : -1e30f;
    }
  }
  __syncthreads();

  // ---- custom masked normalization (4 threads per row) ----
  {
    int row = tid >> 2, sub = tid & 3;
    int tok = q0 + row;
    if (tok < NTOK) {
      float mx = -1e30f;
      for (int c = sub; c < NPAD; c += 4) mx = fmaxf(mx, S[row * NPAD + c]);
      mx = fmaxf(mx, __shfl_xor(mx, 1));
      mx = fmaxf(mx, __shfl_xor(mx, 2));
      const float* mrow = mask + ((long long)b * NTOK + tok) * NTOK;
      float sum = 0.f;
      for (int c = sub; c < NPAD; c += 4) {
        float e = 0.f;
        if (c < NTOK) e = __expf(S[row * NPAD + c] - mx) * mrow[c];
        S[row * NPAD + c] = e;   // keep e for the normalize pass
        sum += e;
      }
      sum += __shfl_xor(sum, 1);
      sum += __shfl_xor(sum, 2);
      float inv = 1.0f / (sum + EPSC);
      for (int c = sub; c < NPAD; c += 4) {
        float p = (c < NTOK) ? (S[row * NPAD + c] + EPSN) * inv : 0.f;
        P[row * NPAD + c] = f32_bf16(p);
      }
    } else {
      for (int c = sub; c < NPAD; c += 4) P[row * NPAD + c] = 0;
    }
  }
  __syncthreads();

  // ---- pruning-score column sums (keys 1..512, queries 1..512) ----
  for (int key = 1 + tid; key < NTOK; key += 128) {
    float cs = 0.f;
#pragma unroll 4
    for (int r = 0; r < 32; ++r) {
      int tk = q0 + r;
      if (tk >= 1 && tk < NTOK) cs += bf16_f32(P[r * NPAD + key]);
    }
    atomicAdd(&score[b * NKEY + key - 1], cs);
  }

  // ---- O = P @ V (waves split the 64-wide d dimension) ----
  int dw = wave * 16;
  f32x8 o0 = zero8(), o1 = zero8();
  for (int ks = 0; ks < NPAD / 32; ++ks) {
    int key0 = ks * 32;
    bf16x16 PA0 = load_frag(P + (r15) * NPAD + key0 + off);
    bf16x16 PA1 = load_frag(P + (16 + r15) * NPAD + key0 + off);
    bf16x16 VB  = load_frag(vbase + (dw + r15) * NPAD + key0 + off);
    o0 = wmma_bf16(PA0, VB, o0);
    o1 = wmma_bf16(PA1, VB, o1);
  }
  int cc = h * 64 + dw + r15;
#pragma unroll
  for (int r = 0; r < 8; ++r) {
    int t0 = q0 + r + rhi;
    int t1 = t0 + 16;
    if (t0 < NTOK) ob[((long long)b * NTOK + t0) * CDIM + cc] = f32_bf16(o0[r]);
    if (t1 < NTOK) ob[((long long)b * NTOK + t1) * CDIM + cc] = f32_bf16(o1[r]);
  }
}

// ---------------- output projection + bias (32x64 tile / wave) ----------------
__global__ __launch_bounds__(32)
void kproj(const unsigned short* __restrict__ obf,
           const unsigned short* __restrict__ wbf,
           const float* __restrict__ bias,
           float* __restrict__ out) {
  const int lane = threadIdx.x;
  const int r15  = lane & 15;
  const int off  = (lane >> 4) * 8;
  const int m0 = blockIdx.x * 32;
  const int n0 = blockIdx.y * 64;

  const unsigned short* a0 = obf + (m0 + r15) * CDIM;
  const unsigned short* a1 = a0 + 16 * CDIM;
  const unsigned short* bp = wbf + (n0 + r15) * CDIM;

  f32x8 acc[8];
#pragma unroll
  for (int i = 0; i < 8; ++i) acc[i] = zero8();

#pragma unroll 2
  for (int k0 = 0; k0 < CDIM; k0 += 32) {
    bf16x16 A0 = load_frag(a0 + k0 + off);
    bf16x16 A1 = load_frag(a1 + k0 + off);
#pragma unroll
    for (int nt = 0; nt < 4; ++nt) {
      bf16x16 B = load_frag(bp + nt * 16 * CDIM + k0 + off);
      acc[nt]     = wmma_bf16(A0, B, acc[nt]);
      acc[4 + nt] = wmma_bf16(A1, B, acc[4 + nt]);
    }
  }

  const int rhi = 8 * (lane >> 4);
#pragma unroll
  for (int mt = 0; mt < 2; ++mt) {
#pragma unroll
    for (int nt = 0; nt < 4; ++nt) {
      const f32x8& a = acc[mt * 4 + nt];
#pragma unroll
      for (int r = 0; r < 8; ++r) {
        int m = m0 + mt * 16 + r + rhi;
        int n = n0 + nt * 16 + r15;
        out[(long long)m * CDIM + n] = a[r] + bias[n];
      }
    }
  }
}

// ---------------- per-batch top-k via bitonic sort of 512 keys ----------------
__global__ __launch_bounds__(256)
void ktopk(const float* __restrict__ score, int* __restrict__ outIdx) {
  __shared__ unsigned long long keys[NKEY];
  int b = blockIdx.x, tid = threadIdx.x;
  for (int i = tid; i < NKEY; i += 256) {
    unsigned int u = __float_as_uint(score[b * NKEY + i]);
    u = (u & 0x80000000u) ? ~u : (u | 0x80000000u);  // monotonic float->uint
    // ascending sort on (~u, idx) == descending on value, ties -> smaller idx
    keys[i] = ((unsigned long long)(~u) << 32) | (unsigned int)i;
  }
  __syncthreads();
  for (int k = 2; k <= NKEY; k <<= 1) {
    for (int j = k >> 1; j > 0; j >>= 1) {
      for (int i = tid; i < NKEY; i += 256) {
        int ixj = i ^ j;
        if (ixj > i) {
          bool up = ((i & k) == 0);
          unsigned long long a = keys[i], c = keys[ixj];
          if ((a > c) == up) { keys[i] = c; keys[ixj] = a; }
        }
      }
      __syncthreads();
    }
  }
  for (int i = tid; i < NLEFT; i += 256)
    outIdx[b * NLEFT + i] = (int)(keys[i] & 0xFFFFFFFFu);
}

// ---------------- host launcher ----------------
extern "C" void kernel_launch(void* const* d_in, const int* in_sizes, int n_in,
                              void* d_out, int out_size, void* d_ws, size_t ws_size,
                              hipStream_t stream) {
  const float* x     = (const float*)d_in[0];
  const float* mask  = (const float*)d_in[1];
  const float* wqkv  = (const float*)d_in[2];
  const float* wproj = (const float*)d_in[3];
  const float* bproj = (const float*)d_in[4];
  float* out = (float*)d_out;
  int*   tidx = (int*)(out + (size_t)MROWS * CDIM);

  char* ws = (char*)d_ws;
  size_t o = 0;
  auto alloc = [&](size_t bytes) -> char* {
    char* p = ws + o;
    o += (bytes + 255) & ~(size_t)255;
    return p;
  };
  unsigned short* xbf   = (unsigned short*)alloc((size_t)MROWS * CDIM * 2);
  unsigned short* wqbf  = (unsigned short*)alloc((size_t)THREEC * CDIM * 2);
  unsigned short* wpbf  = (unsigned short*)alloc((size_t)CDIM * CDIM * 2);
  unsigned short* qb    = (unsigned short*)alloc((size_t)BATCH * NH * NPAD * HDIM * 2);
  unsigned short* kb    = (unsigned short*)alloc((size_t)BATCH * NH * NPAD * HDIM * 2);
  unsigned short* vtb   = (unsigned short*)alloc((size_t)BATCH * NH * HDIM * NPAD * 2);
  unsigned short* obf   = (unsigned short*)alloc((size_t)MROWS * CDIM * 2);
  float*          score = (float*)alloc((size_t)BATCH * NKEY * 4);

  // zero the padded q/k/vT region (contiguous) and the score accumulator
  long long qkvBytes = 3LL * BATCH * NH * NPAD * HDIM * 2;
  kzero<<<2048, 256, 0, stream>>>((float4*)qb, qkvBytes / 16);
  kzero<<<16, 256, 0, stream>>>((float4*)score, (long long)BATCH * NKEY * 4 / 16);

  // fp32 -> bf16 staging
  kconv<<<4096, 256, 0, stream>>>(x, xbf, MROWS * CDIM);
  kconv<<<1024, 256, 0, stream>>>(wqkv, wqbf, THREEC * CDIM);
  kconv<<<512, 256, 0, stream>>>(wproj, wpbf, CDIM * CDIM);

  // qkv projection (32x64 tiles)
  kqkv<<<dim3(MROWS / 32, THREEC / 64), 32, 0, stream>>>(xbf, wqbf, qb, kb, vtb);

  // attention + pruning-score accumulation
  size_t smem = (size_t)32 * NPAD * 4 + (size_t)32 * NPAD * 2 + (size_t)32 * HDIM * 2;
  kattn<<<BATCH * NH * QTILES, 128, smem, stream>>>(qb, kb, vtb, mask, obf, score);

  // output projection (32x64 tiles)
  kproj<<<dim3(MROWS / 32, CDIM / 64), 32, 0, stream>>>(obf, wpbf, bproj, out);

  // top-k indices
  ktopk<<<BATCH, 256, 0, stream>>>(score, tidx);

  (void)in_sizes; (void)n_in; (void)out_size; (void)ws_size;
}
